// StandardisedSignatureHedger_51848845197657
// MI455X (gfx1250) — matrix operand order repaired and checked by reading, not compiled, not run on hardware
//
#include <hip/hip_runtime.h>
#include <hip/hip_bf16.h>
#include <math.h>

// Problem constants (from reference)
#define B_TOT   16384
#define N_STEPS 250
#define H_DIM   256
#define F_DIM   8
#define BM      64            // paths per workgroup
#define LDA     264           // padded LDS row stride in halfs (keeps 16B align)
#define NEG_SLOPE 0.2f
#define S0_C    100.0f
#define T_C     1.0f
#define SIGMA_C 0.235f

typedef __attribute__((ext_vector_type(16))) _Float16 v16h;
typedef __attribute__((ext_vector_type(8)))  _Float16 v8h;
typedef __attribute__((ext_vector_type(8)))  float    v8f;
typedef __attribute__((ext_vector_type(2)))  float    v2f;

__device__ __forceinline__ float lrelu(float v) { return fmaxf(v, NEG_SLOPE * v); }

// ---- prep: convert Wr1[2][256][256], Wr2[2][256][256] (f32) to f16 in ws ----
// layout in ws: mat0=Wr1[0], mat1=Wr2[0], mat2=Wr1[1], mat3=Wr2[1]; each [n][k]
__global__ void prep_weights_kernel(const float* __restrict__ Wr1,
                                    const float* __restrict__ Wr2,
                                    _Float16* __restrict__ W16) {
  int idx = blockIdx.x * blockDim.x + threadIdx.x;   // 0 .. 4*65536-1
  int mat = idx >> 16;
  int within = idx & 0xFFFF;
  const float* src = (mat & 1) ? Wr2 : Wr1;
  W16[idx] = (_Float16)src[(mat >> 1) * (H_DIM * H_DIM) + within];
}

// ---- persistent per-batch-tile time-loop kernel ----
// Wave mapping: wave w owns ALL 4 M-tiles (64 rows) and 2 N-tiles (cols w*32..w*32+31)
// => every weight byte is loaded exactly once per workgroup per GEMM.
__global__ __launch_bounds__(256)
void hedger_kernel(const float* __restrict__ S,
                   const float* __restrict__ feat_mean,
                   const float* __restrict__ feat_std,
                   const float* __restrict__ W_in,
                   const float* __restrict__ b_in,
                   const float* __restrict__ br1,
                   const float* __restrict__ br2,
                   const float* __restrict__ W_out,
                   const float* __restrict__ b_out,
                   const _Float16* __restrict__ W16,
                   float* __restrict__ out) {
  __shared__ __align__(16) _Float16 hA[BM * LDA];   // activation (A operand), f16
  __shared__ __align__(16) _Float16 uA[BM * LDA];   // inner activation, f16
  __shared__ float zL[BM][F_DIM];
  __shared__ float red[BM * 128];                   // output-head partials
  __shared__ float red2[4 * BM];                    // phase-2 partials
  __shared__ float deltaBuf[BM];
  __shared__ float WinL[H_DIM * F_DIM];
  __shared__ float binL[H_DIM];
  __shared__ float br1L[2][H_DIM];
  __shared__ float br2L[2][H_DIM];
  __shared__ float WoutL[H_DIM];
  __shared__ float meanL[F_DIM];
  __shared__ float invstdL[F_DIM];

  const int tid  = threadIdx.x;
  const int lane = tid & 31;
  const int wave = tid >> 5;
  const int l15  = lane & 15;
  const int lhi  = lane >> 4;
  const int nw   = wave * 32;          // wave's 32-column strip
  const int b0   = blockIdx.x * BM;

  // preload constants into LDS
  for (int i = tid; i < H_DIM * F_DIM; i += 256) WinL[i] = W_in[i];
  for (int i = tid; i < H_DIM; i += 256) {
    binL[i]    = b_in[i];
    br1L[0][i] = br1[i];        br1L[1][i] = br1[H_DIM + i];
    br2L[0][i] = br2[i];        br2L[1][i] = br2[H_DIM + i];
    WoutL[i]   = W_out[i];
  }
  if (tid < F_DIM) { meanL[tid] = feat_mean[tid]; invstdL[tid] = 1.0f / feat_std[tid]; }
  if (tid < BM) deltaBuf[tid] = 0.0f;
  __syncthreads();

  // per-path feature-scan state (threads 0..63)
  float xcur = 0.f, accRv = 0.f, accArea = 0.f, rmin = INFINITY;
  if (tid < BM) xcur = __logf(S[(size_t)(b0 + tid) * (N_STEPS + 1)] * (1.0f / S0_C));

  const float dt      = T_C / (float)N_STEPS;
  const float inv_xi0 = 1.0f / (SIGMA_C * SIGMA_C);
  const float bout    = b_out[0];

  v8f h_reg[4][2];   // f32 residual h: [Mtile][Ntile-in-strip], 8 rows/lane each

  for (int k = 0; k < N_STEPS; ++k) {
    // ---- stage 0a: standardized features (scan threads) ----
    if (tid < BM) {
      rmin = fminf(rmin, xcur);
      float t   = k * dt;
      float tau = T_C - t;
      float f[8];
      f[0] = xcur;            f[1] = tau;
      f[2] = sqrtf(tau);      f[3] = deltaBuf[tid];      // delta_prev (mean=0,std=1)
      f[4] = accRv * inv_xi0; f[5] = accArea;
      f[6] = rmin;            f[7] = t * (1.0f / T_C);
      #pragma unroll
      for (int q = 0; q < F_DIM; ++q) zL[tid][q] = (f[q] - meanL[q]) * invstdL[q];
      // advance scan (S has N+1 columns, k+1 <= 250 always valid)
      float xn = __logf(S[(size_t)(b0 + tid) * (N_STEPS + 1) + k + 1] * (1.0f / S0_C));
      float dx = xn - xcur;
      accRv   += dx * dx;
      accArea += xcur * dx;
      xcur = xn;
    }
    __syncthreads();

    // ---- stage 0b: input layer h = lrelu(z @ W_in^T + b_in) via f32 K=4 WMMA ----
    #pragma unroll
    for (int mi = 0; mi < 4; ++mi) {
      #pragma unroll
      for (int jj = 0; jj < 2; ++jj) {
        int n = nw + jj * 16 + l15;
        float bv = binL[n];
        v8f c;
        #pragma unroll
        for (int r = 0; r < 8; ++r) c[r] = bv;
        int m = mi * 16 + l15;
        #pragma unroll
        for (int ko = 0; ko < 8; ko += 4) {       // K = 8 as two K=4 WMMAs
          v2f a, b;
          a[0] = zL[m][ko + lhi * 2 + 0];
          a[1] = zL[m][ko + lhi * 2 + 1];
          b[0] = WinL[n * F_DIM + ko + lhi * 2 + 0];
          b[1] = WinL[n * F_DIM + ko + lhi * 2 + 1];
          c = __builtin_amdgcn_wmma_f32_16x16x4_f32(false, a, false, b,
                                                    (short)0, c, false, false);
        }
        #pragma unroll
        for (int r = 0; r < 8; ++r) {
          float v = lrelu(c[r]);
          c[r] = v;
          hA[(mi * 16 + lhi * 8 + r) * LDA + n] = (_Float16)v;
        }
        h_reg[mi][jj] = c;
      }
    }
    __syncthreads();

    // ---- residual blocks: two 256x256 WMMA GEMMs each ----
    for (int blk = 0; blk < 2; ++blk) {
      // GEMM1: u = lrelu(h @ Wr1^T + br1)
      const _Float16* Wm1 = W16 + (size_t)(2 * blk) * (H_DIM * H_DIM);
      v8f acc[4][2];
      #pragma unroll
      for (int jj = 0; jj < 2; ++jj) {
        float bv = br1L[blk][nw + jj * 16 + l15];
        #pragma unroll
        for (int mi = 0; mi < 4; ++mi)
          #pragma unroll
          for (int r = 0; r < 8; ++r) acc[mi][jj][r] = bv;
      }
      #pragma unroll 2
      for (int kt = 0; kt < 8; ++kt) {
        int c0 = kt * 32 + lhi * 16;
        v16h af[4];
        #pragma unroll
        for (int mi = 0; mi < 4; ++mi) {
          int row = mi * 16 + l15;
          v8h alo = *(const v8h*)&hA[row * LDA + c0];
          v8h ahi = *(const v8h*)&hA[row * LDA + c0 + 8];
          af[mi] = __builtin_shufflevector(alo, ahi, 0,1,2,3,4,5,6,7,8,9,10,11,12,13,14,15);
        }
        #pragma unroll
        for (int jj = 0; jj < 2; ++jj) {
          int n = nw + jj * 16 + l15;
          v16h bf = *(const v16h*)&Wm1[(size_t)n * H_DIM + kt * 32 + lhi * 16];
          #pragma unroll
          for (int mi = 0; mi < 4; ++mi)
            acc[mi][jj] = __builtin_amdgcn_wmma_f32_16x16x32_f16(false, af[mi], false, bf,
                                                                 (short)0, acc[mi][jj], false, false);
        }
      }
      #pragma unroll
      for (int mi = 0; mi < 4; ++mi)
        #pragma unroll
        for (int jj = 0; jj < 2; ++jj) {
          int n = nw + jj * 16 + l15;
          #pragma unroll
          for (int r = 0; r < 8; ++r) {
            float v = lrelu(acc[mi][jj][r]);
            uA[(mi * 16 + lhi * 8 + r) * LDA + n] = (_Float16)v;
          }
        }
      __syncthreads();

      // GEMM2: h = h + u @ Wr2^T + br2  (C preloaded with h + bias)
      const _Float16* Wm2 = W16 + (size_t)(2 * blk + 1) * (H_DIM * H_DIM);
      #pragma unroll
      for (int jj = 0; jj < 2; ++jj) {
        float bv = br2L[blk][nw + jj * 16 + l15];
        #pragma unroll
        for (int mi = 0; mi < 4; ++mi)
          #pragma unroll
          for (int r = 0; r < 8; ++r) h_reg[mi][jj][r] += bv;
      }
      #pragma unroll 2
      for (int kt = 0; kt < 8; ++kt) {
        int c0 = kt * 32 + lhi * 16;
        v16h af[4];
        #pragma unroll
        for (int mi = 0; mi < 4; ++mi) {
          int row = mi * 16 + l15;
          v8h alo = *(const v8h*)&uA[row * LDA + c0];
          v8h ahi = *(const v8h*)&uA[row * LDA + c0 + 8];
          af[mi] = __builtin_shufflevector(alo, ahi, 0,1,2,3,4,5,6,7,8,9,10,11,12,13,14,15);
        }
        #pragma unroll
        for (int jj = 0; jj < 2; ++jj) {
          int n = nw + jj * 16 + l15;
          v16h bf = *(const v16h*)&Wm2[(size_t)n * H_DIM + kt * 32 + lhi * 16];
          #pragma unroll
          for (int mi = 0; mi < 4; ++mi)
            h_reg[mi][jj] = __builtin_amdgcn_wmma_f32_16x16x32_f16(false, af[mi], false, bf,
                                                                   (short)0, h_reg[mi][jj], false, false);
        }
      }
      // publish new h as f16 A operand for the NEXT block only; after the last
      // block h is consumed from registers by the output head.
      if (blk == 0) {
        #pragma unroll
        for (int mi = 0; mi < 4; ++mi)
          #pragma unroll
          for (int jj = 0; jj < 2; ++jj) {
            int n = nw + jj * 16 + l15;
            #pragma unroll
            for (int r = 0; r < 8; ++r)
              hA[(mi * 16 + lhi * 8 + r) * LDA + n] = (_Float16)h_reg[mi][jj][r];
          }
      }
      __syncthreads();
    }

    // ---- output head: delta = sigmoid(h @ W_out^T + b_out) ----
    {
      float wv0 = WoutL[nw + l15];
      float wv1 = WoutL[nw + 16 + l15];
      #pragma unroll
      for (int mi = 0; mi < 4; ++mi)
        #pragma unroll
        for (int r = 0; r < 8; ++r) {
          float p = fmaf(h_reg[mi][0][r], wv0, h_reg[mi][1][r] * wv1);
          red[(mi * 16 + lhi * 8 + r) * 128 + wave * 16 + l15] = p;
        }
    }
    __syncthreads();

    // phase 1: all 256 threads, each sums 32 of a row's 128 partials
    {
      int row = tid & (BM - 1);
      int q   = tid >> 6;           // 0..3
      float s = 0.f;
      #pragma unroll 8
      for (int c = 0; c < 32; ++c) s += red[row * 128 + q * 32 + c];
      red2[q * BM + row] = s;
    }
    __syncthreads();

    if (tid < BM) {
      float s = bout + red2[tid] + red2[BM + tid] + red2[2 * BM + tid] + red2[3 * BM + tid];
      float d = 1.0f / (1.0f + __expf(-s));
      out[(size_t)(b0 + tid) * N_STEPS + k] = d;
      deltaBuf[tid] = d;
    }
    __syncthreads();
  }
}

extern "C" void kernel_launch(void* const* d_in, const int* in_sizes, int n_in,
                              void* d_out, int out_size, void* d_ws, size_t ws_size,
                              hipStream_t stream) {
  (void)in_sizes; (void)n_in; (void)out_size; (void)ws_size;
  const float* S         = (const float*)d_in[0];
  const float* feat_mean = (const float*)d_in[1];
  const float* feat_std  = (const float*)d_in[2];
  const float* W_in      = (const float*)d_in[3];
  const float* b_in      = (const float*)d_in[4];
  const float* Wr1       = (const float*)d_in[5];
  const float* br1       = (const float*)d_in[6];
  const float* Wr2       = (const float*)d_in[7];
  const float* br2       = (const float*)d_in[8];
  const float* W_out     = (const float*)d_in[9];
  const float* b_out     = (const float*)d_in[10];

  _Float16* W16 = (_Float16*)d_ws;     // 4 * 256 * 256 * 2B = 512 KB scratch
  float* out = (float*)d_out;

  prep_weights_kernel<<<(4 * H_DIM * H_DIM) / 256, 256, 0, stream>>>(Wr1, Wr2, W16);
  hedger_kernel<<<B_TOT / BM, 256, 0, stream>>>(S, feat_mean, feat_std, W_in, b_in,
                                                br1, br2, W_out, b_out, W16, out);
}
